// Attention_34153579938524
// MI455X (gfx1250) — matrix operand order repaired and checked by reading, not compiled, not run on hardware
//
#include <hip/hip_runtime.h>

// ---------------- CDNA5 (gfx1250) wave32 WMMA plumbing ----------------
typedef __bf16 bf16;
typedef bf16           v16bf __attribute__((ext_vector_type(16)));
typedef float          v8f   __attribute__((ext_vector_type(8)));
typedef unsigned int   v4u   __attribute__((ext_vector_type(4)));
typedef float          v4f   __attribute__((ext_vector_type(4)));
typedef unsigned short v4s   __attribute__((ext_vector_type(4)));

struct U8x32 { v4u a, b; };  // 32 bytes == v16bf

static __device__ __forceinline__ v16bf make_frag(v4u lo, v4u hi) {
  U8x32 t; t.a = lo; t.b = hi;
  return __builtin_bit_cast(v16bf, t);
}
static __device__ __forceinline__ v8f zero8() {
  v8f z;
#pragma unroll
  for (int i = 0; i < 8; ++i) z[i] = 0.0f;
  return z;
}
static __device__ __forceinline__ unsigned short f2bf(float f) {
  unsigned int u = __float_as_uint(f);
  u += 0x7fffu + ((u >> 16) & 1u);   // round-to-nearest-even
  return (unsigned short)(u >> 16);
}
static __device__ __forceinline__ v8f wmma_bf16(v16bf a, v16bf b, v8f c) {
  return __builtin_amdgcn_wmma_f32_16x16x32_bf16(false, a, false, b, (short)0, c,
                                                 false, false);
}

// --- CDNA5 async global->LDS copy (ASYNCcnt-tracked, ISA ch.10) -------------
// LDS byte offset = low 32 bits of the generic pointer (aperture is in [63:32]).
static __device__ __forceinline__ unsigned lds_lo32(const void* p) {
  return (unsigned)(unsigned long long)p;
}
static __device__ __forceinline__ void async_ld_b128(const void* gp,
                                                     unsigned lds_addr) {
  asm volatile("global_load_async_to_lds_b128 %0, %1, off"
               :: "v"(lds_addr), "v"(gp)
               : "memory");
}
static __device__ __forceinline__ void wait_async0() {
  asm volatile("s_wait_asynccnt 0x0" ::: "memory");
}

#define EMBED 1024
#define HEADS 16
#define HD    64
#define BATCH 8
#define SEQ   1024
#define MTOT  (BATCH * SEQ)  // 8192

// ---------------- fp32 -> bf16 convert (x4 vectorized, sizes % 1024 == 0) ---
__global__ void cvt_f32_bf16(const float* __restrict__ src,
                             unsigned short* __restrict__ dst) {
  int i = (blockIdx.x * blockDim.x + threadIdx.x) * 4;
  v4f f = *(const v4f*)(src + i);
  v4s o;
#pragma unroll
  for (int j = 0; j < 4; ++j) o[j] = f2bf(f[j]);
  *(v4s*)(dst + i) = o;
}

// ---------------- QKV GEMM: [8192x1024] x [1024x3072] -> Q/K/V bf16 --------
// Block: 256 threads (8 waves), tile 128x128, K-step 32.
__global__ __launch_bounds__(256) void gemm_qkv(
    const unsigned short* __restrict__ A,   // x  bf16 [8192 x 1024]
    const unsigned short* __restrict__ Bm,  // w  bf16 [1024 x 3072]
    unsigned short* __restrict__ Qb,        // [B*H, SEQ, HD]
    unsigned short* __restrict__ Kb,
    unsigned short* __restrict__ Vb) {
  constexpr int K = EMBED, N = 3 * EMBED;
  __shared__ unsigned short sA[128 * 40];  // [m][k], k-stride 40 (pad)
  __shared__ unsigned short sB[128 * 40];  // transposed [n][k], k-stride 40

  const int tid = threadIdx.x;
  const int lane = tid & 31, wv = tid >> 5;
  const int hf = lane >> 4, ln = lane & 15;
  const int wm = (wv >> 1) * 32, wn = (wv & 1) * 64;
  const int bm = blockIdx.y * 128, bn = blockIdx.x * 128;

  v8f acc[2][4];
#pragma unroll
  for (int mi = 0; mi < 2; ++mi)
#pragma unroll
    for (int ni = 0; ni < 4; ++ni) acc[mi][ni] = zero8();

  for (int kt = 0; kt < K; kt += 32) {
    // stage A tile 128x32: async global->LDS b128 (no VGPR round-trip)
#pragma unroll
    for (int i = 0; i < 2; ++i) {
      int c = tid * 2 + i;               // 0..511
      int r = c >> 2, cc = (c & 3) << 3;
      async_ld_b128(A + (bm + r) * K + kt + cc, lds_lo32(&sA[r * 40 + cc]));
    }
    // stage B tile 32x128, transposed into [n][k] (element reshuffle -> VALU)
#pragma unroll
    for (int i = 0; i < 2; ++i) {
      int c = tid * 2 + i;
      int kr = c >> 4, nc = (c & 15) << 3;
      v4u d = *(const v4u*)(Bm + (kt + kr) * N + bn + nc);
#pragma unroll
      for (int j = 0; j < 8; ++j) {
        unsigned int w = d[j >> 1];
        sB[(nc + j) * 40 + kr] =
            (unsigned short)((j & 1) ? (w >> 16) : (w & 0xffffu));
      }
    }
    if (kt + 32 < K)  // WGP-scope prefetch of the next K tile
      __builtin_prefetch(A + (bm + (tid >> 1)) * K + kt + 32, 0, 3);
    wait_async0();
    __syncthreads();

    v16bf af[2], bfr[4];
#pragma unroll
    for (int mi = 0; mi < 2; ++mi) {
      int row = wm + mi * 16 + ln;
      af[mi] = make_frag(*(const v4u*)(&sA[row * 40 + hf * 8]),
                         *(const v4u*)(&sA[row * 40 + 16 + hf * 8]));
    }
#pragma unroll
    for (int ni = 0; ni < 4; ++ni) {
      int col = wn + ni * 16 + ln;
      bfr[ni] = make_frag(*(const v4u*)(&sB[col * 40 + hf * 8]),
                          *(const v4u*)(&sB[col * 40 + 16 + hf * 8]));
    }
#pragma unroll
    for (int mi = 0; mi < 2; ++mi)
#pragma unroll
      for (int ni = 0; ni < 4; ++ni)
        acc[mi][ni] = wmma_bf16(af[mi], bfr[ni], acc[mi][ni]);
    __syncthreads();
  }

  // epilogue: scatter to Q/K/V laid out [(b*H+h), n, d]
#pragma unroll
  for (int ni = 0; ni < 4; ++ni) {
    int gcol = bn + wn + ni * 16 + ln;
    int sel = gcol >> 10;
    int rem = gcol & 1023;
    int h = rem >> 6, d = rem & 63;
    unsigned short* dst = (sel == 0) ? Qb : ((sel == 1) ? Kb : Vb);
#pragma unroll
    for (int mi = 0; mi < 2; ++mi)
#pragma unroll
      for (int v = 0; v < 8; ++v) {
        int grow = bm + wm + mi * 16 + v + 8 * hf;
        int b = grow >> 10, np = grow & 1023;
        dst[(((b * HEADS + h) * SEQ + np) * HD) + d] = f2bf(acc[mi][ni][v]);
      }
  }
}

// ---------------- fused flash attention per (b,h) ---------------------------
// Block: 128 threads (4 waves). Each wave: 16 query rows, O = 16x64 fp32.
__global__ __launch_bounds__(128) void attn_fused(
    const unsigned short* __restrict__ Qb, const unsigned short* __restrict__ Kb,
    const unsigned short* __restrict__ Vb,
    unsigned short* __restrict__ Ob) {  // bf16 [B, SEQ, EMBED]
  __shared__ unsigned short sK[32 * 72];      // [key][d], d-stride 72
  __shared__ unsigned short sV[64 * 40];      // transposed [d][key], stride 40
  __shared__ unsigned short sP[4][16 * 40];   // per-wave P tile [m][k]

  const int tid = threadIdx.x;
  const int lane = tid & 31, wv = tid >> 5;
  const int hf = lane >> 4, ln = lane & 15;
  const int bh = blockIdx.y;
  const int q0 = blockIdx.x * 64 + wv * 16;
  const unsigned short* Qp = Qb + (size_t)bh * SEQ * HD;
  const unsigned short* Kp = Kb + (size_t)bh * SEQ * HD;
  const unsigned short* Vp = Vb + (size_t)bh * SEQ * HD;
  unsigned short* sPw = sP[wv];

  // Q resident in registers: two 16x32 A-fragments over d
  v16bf qf[2];
#pragma unroll
  for (int di = 0; di < 2; ++di) {
    int row = q0 + ln;
    qf[di] = make_frag(*(const v4u*)(Qp + row * HD + di * 32 + hf * 8),
                       *(const v4u*)(Qp + row * HD + di * 32 + 16 + hf * 8));
  }

  v8f o[4];
#pragma unroll
  for (int ni = 0; ni < 4; ++ni) o[ni] = zero8();
  float mrow[8], lrow[8];
#pragma unroll
  for (int v = 0; v < 8; ++v) { mrow[v] = -3.0e38f; lrow[v] = 0.0f; }
  const float scale = 0.125f;  // 1/sqrt(64)

  for (int j = 0; j < SEQ; j += 32) {
    __syncthreads();
    // stage K via async global->LDS; V needs a transpose so it goes via VGPRs
#pragma unroll
    for (int i = 0; i < 2; ++i) {
      int c = tid * 2 + i;  // 0..255
      int kr = c >> 3, dc = (c & 7) << 3;
      async_ld_b128(Kp + (j + kr) * HD + dc, lds_lo32(&sK[kr * 72 + dc]));
      v4u d = *(const v4u*)(Vp + (j + kr) * HD + dc);
#pragma unroll
      for (int t = 0; t < 8; ++t) {
        unsigned int w = d[t >> 1];
        sV[(dc + t) * 40 + kr] =
            (unsigned short)((t & 1) ? (w >> 16) : (w & 0xffffu));
      }
    }
    wait_async0();
    __syncthreads();

    // S = Q K^T : two 16-key groups, contraction over d in two WMMA steps
    v8f s[2];
#pragma unroll
    for (int kk = 0; kk < 2; ++kk) {
      v8f c = zero8();
#pragma unroll
      for (int di = 0; di < 2; ++di) {
        int key = kk * 16 + ln;
        v16bf kf =
            make_frag(*(const v4u*)(&sK[key * 72 + di * 32 + hf * 8]),
                      *(const v4u*)(&sK[key * 72 + di * 32 + 16 + hf * 8]));
        c = wmma_bf16(qf[di], kf, c);
      }
      s[kk] = c;
    }

    // online softmax: row m = v + 8*hf lives across the 16 lanes of this half
#pragma unroll
    for (int v = 0; v < 8; ++v) {
      float s0 = s[0][v] * scale;
      float s1 = s[1][v] * scale;
      float mx = fmaxf(s0, s1);
#pragma unroll
      for (int off = 1; off < 16; off <<= 1)
        mx = fmaxf(mx, __shfl_xor(mx, off, 32));
      float mnew = fmaxf(mrow[v], mx);
      float alpha = __expf(mrow[v] - mnew);
      float p0 = __expf(s0 - mnew);
      float p1 = __expf(s1 - mnew);
      float rs = p0 + p1;
#pragma unroll
      for (int off = 1; off < 16; off <<= 1) rs += __shfl_xor(rs, off, 32);
      mrow[v] = mnew;
      lrow[v] = lrow[v] * alpha + rs;
#pragma unroll
      for (int ni = 0; ni < 4; ++ni) o[ni][v] *= alpha;
      int m = v + 8 * hf;
      sPw[m * 40 + ln] = f2bf(p0);        // C-layout -> A-layout via LDS
      sPw[m * 40 + 16 + ln] = f2bf(p1);
    }
    __builtin_amdgcn_wave_barrier();  // keep DS store->load order (same wave)

    // O += P V : P as 16x32 A-frag from LDS, V^T B-frags from LDS
    v16bf pf = make_frag(*(const v4u*)(&sPw[ln * 40 + hf * 8]),
                         *(const v4u*)(&sPw[ln * 40 + 16 + hf * 8]));
#pragma unroll
    for (int ni = 0; ni < 4; ++ni) {
      int dcol = ni * 16 + ln;
      v16bf vf = make_frag(*(const v4u*)(&sV[dcol * 40 + hf * 8]),
                           *(const v4u*)(&sV[dcol * 40 + 16 + hf * 8]));
      o[ni] = wmma_bf16(pf, vf, o[ni]);
    }
  }

  // normalize and store to bf16 [b, n, h*64 + d]
  const int b = bh >> 4, h = bh & 15;
#pragma unroll
  for (int ni = 0; ni < 4; ++ni) {
    int d = h * HD + ni * 16 + ln;
#pragma unroll
    for (int v = 0; v < 8; ++v) {
      int m = q0 + v + 8 * hf;
      Ob[((size_t)b * SEQ + m) * EMBED + d] = f2bf(o[ni][v] / lrow[v]);
    }
  }
}

// ---------------- proj GEMM + bias: [8192x1024] x [1024x1024] -> f32 --------
__global__ __launch_bounds__(256) void gemm_proj(
    const unsigned short* __restrict__ A,   // attn out bf16 [8192 x 1024]
    const unsigned short* __restrict__ Bm,  // w_proj bf16 [1024 x 1024]
    const float* __restrict__ bias,         // [1024]
    float* __restrict__ Out) {              // [8192 x 1024] f32
  constexpr int K = EMBED, N = EMBED;
  __shared__ unsigned short sA[128 * 40];
  __shared__ unsigned short sB[128 * 40];

  const int tid = threadIdx.x;
  const int lane = tid & 31, wv = tid >> 5;
  const int hf = lane >> 4, ln = lane & 15;
  const int wm = (wv >> 1) * 32, wn = (wv & 1) * 64;
  const int bm = blockIdx.y * 128, bn = blockIdx.x * 128;

  v8f acc[2][4];
#pragma unroll
  for (int mi = 0; mi < 2; ++mi)
#pragma unroll
    for (int ni = 0; ni < 4; ++ni) acc[mi][ni] = zero8();

  for (int kt = 0; kt < K; kt += 32) {
#pragma unroll
    for (int i = 0; i < 2; ++i) {
      int c = tid * 2 + i;
      int r = c >> 2, cc = (c & 3) << 3;
      async_ld_b128(A + (bm + r) * K + kt + cc, lds_lo32(&sA[r * 40 + cc]));
    }
#pragma unroll
    for (int i = 0; i < 2; ++i) {
      int c = tid * 2 + i;
      int kr = c >> 4, nc = (c & 15) << 3;
      v4u d = *(const v4u*)(Bm + (kt + kr) * N + bn + nc);
#pragma unroll
      for (int j = 0; j < 8; ++j) {
        unsigned int w = d[j >> 1];
        sB[(nc + j) * 40 + kr] =
            (unsigned short)((j & 1) ? (w >> 16) : (w & 0xffffu));
      }
    }
    if (kt + 32 < K)
      __builtin_prefetch(A + (bm + (tid >> 1)) * K + kt + 32, 0, 3);
    wait_async0();
    __syncthreads();

    v16bf af[2], bfr[4];
#pragma unroll
    for (int mi = 0; mi < 2; ++mi) {
      int row = wm + mi * 16 + ln;
      af[mi] = make_frag(*(const v4u*)(&sA[row * 40 + hf * 8]),
                         *(const v4u*)(&sA[row * 40 + 16 + hf * 8]));
    }
#pragma unroll
    for (int ni = 0; ni < 4; ++ni) {
      int col = wn + ni * 16 + ln;
      bfr[ni] = make_frag(*(const v4u*)(&sB[col * 40 + hf * 8]),
                          *(const v4u*)(&sB[col * 40 + 16 + hf * 8]));
    }
#pragma unroll
    for (int mi = 0; mi < 2; ++mi)
#pragma unroll
      for (int ni = 0; ni < 4; ++ni)
        acc[mi][ni] = wmma_bf16(af[mi], bfr[ni], acc[mi][ni]);
    __syncthreads();
  }

#pragma unroll
  for (int ni = 0; ni < 4; ++ni) {
    int gcol = bn + wn + ni * 16 + ln;
    float bv = bias[gcol];
#pragma unroll
    for (int mi = 0; mi < 2; ++mi)
#pragma unroll
      for (int v = 0; v < 8; ++v) {
        int grow = bm + wm + mi * 16 + v + 8 * hf;
        Out[(size_t)grow * N + gcol] = acc[mi][ni][v] + bv;
      }
  }
}

// ---------------- host-side launcher ----------------------------------------
extern "C" void kernel_launch(void* const* d_in, const int* in_sizes, int n_in,
                              void* d_out, int out_size, void* d_ws,
                              size_t ws_size, hipStream_t stream) {
  (void)in_sizes; (void)n_in; (void)out_size; (void)ws_size;
  const float* x = (const float*)d_in[0];
  const float* w_qkv = (const float*)d_in[1];
  const float* w_proj = (const float*)d_in[2];
  const float* b_proj = (const float*)d_in[3];
  float* out = (float*)d_out;

  char* ws = (char*)d_ws;
  unsigned short* xbf = (unsigned short*)ws;  ws += (size_t)MTOT * EMBED * 2;
  unsigned short* wqk = (unsigned short*)ws;  ws += (size_t)EMBED * 3 * EMBED * 2;
  unsigned short* wpj = (unsigned short*)ws;  ws += (size_t)EMBED * EMBED * 2;
  unsigned short* Qb  = (unsigned short*)ws;  ws += (size_t)MTOT * EMBED * 2;
  unsigned short* Kb  = (unsigned short*)ws;  ws += (size_t)MTOT * EMBED * 2;
  unsigned short* Vb  = (unsigned short*)ws;  ws += (size_t)MTOT * EMBED * 2;
  unsigned short* abf = (unsigned short*)ws;  ws += (size_t)MTOT * EMBED * 2;

  cvt_f32_bf16<<<(MTOT * EMBED) / 1024, 256, 0, stream>>>(x, xbf);
  cvt_f32_bf16<<<(EMBED * 3 * EMBED) / 1024, 256, 0, stream>>>(w_qkv, wqk);
  cvt_f32_bf16<<<(EMBED * EMBED) / 1024, 256, 0, stream>>>(w_proj, wpj);

  gemm_qkv<<<dim3(24, 64), 256, 0, stream>>>(xbf, wqk, Qb, Kb, Vb);
  attn_fused<<<dim3(16, BATCH * HEADS), 128, 0, stream>>>(Qb, Kb, Vb, abf);
  gemm_proj<<<dim3(8, 64), 256, 0, stream>>>(abf, wpj, b_proj, out);
}
